// STHGCN_65317862637908
// MI455X (gfx1250) — matrix operand order repaired and checked by reading, not compiled, not run on hardware
//
#include <hip/hip_runtime.h>
#include <hip/hip_bf16.h>
#include <stdint.h>

// Problem constants (match reference)
#define DD     128
#define NNODES 200000
#define NEDGES 1024
#define NNZV   1600000

// Stage activation tiles into LDS with gfx1250 async copies.
#define USE_ASYNC_TILE 1

typedef __attribute__((ext_vector_type(16))) __bf16 v16bf;
typedef __attribute__((ext_vector_type(8)))  float  v8f;

union Frag16 { v16bf v; unsigned u[8]; };

__device__ __forceinline__ unsigned short f2bf_bits(float f) {
    union { float f; unsigned u; } cv; cv.f = f;
    unsigned u = cv.u;
    unsigned r = u + 0x7FFFu + ((u >> 16) & 1u);   // round-to-nearest-even
    return (unsigned short)(r >> 16);
}

// Async-fill LDS tile: 64x128 bf16 = 16 KB, 1024 16B chunks, 4 per thread.
// The LDS address operand is derived from the tile pointer (ptrtoint) so the
// tile escapes into the asm and the compiler must honor the LDS writes.
// On amdgcn the low 32 bits of a flat shared-aperture address are the LDS
// byte offset, which is exactly what VDST of the async op expects.
__device__ __forceinline__ void stage_tile_64x128(const unsigned short* gsrc,
                                                  unsigned short* tile, int t) {
#if USE_ASYNC_TILE
    unsigned long long gbase = (unsigned long long)(uintptr_t)gsrc;
    unsigned lbase = (unsigned)(uintptr_t)tile;    // LDS byte offset + capture
#pragma unroll
    for (int c = 0; c < 4; ++c) {
        unsigned chunk = (unsigned)t + c * 256u;
        unsigned lds = lbase + chunk * 16u;
        unsigned long long ga = gbase + (unsigned long long)chunk * 16u;
        asm volatile("global_load_async_to_lds_b128 %0, %1, off"
                     :: "v"(lds), "v"(ga) : "memory");
    }
    asm volatile("s_wait_asynccnt 0" ::: "memory");
#else
    const uint4* src = (const uint4*)gsrc;
    uint4* dst = (uint4*)tile;
#pragma unroll
    for (int c = 0; c < 4; ++c) dst[t + c * 256] = src[t + c * 256];
#endif
    __syncthreads();
}

// ---------------- elementwise / prep kernels ----------------

__global__ void k_zero_f32(float* p, int n) {
    int i = blockIdx.x * blockDim.x + threadIdx.x;
    if (i < n) p[i] = 0.0f;
}

__global__ void k_f32_to_bf16(const float* __restrict__ in,
                              unsigned short* __restrict__ out, int n) {
    int i = blockIdx.x * blockDim.x + threadIdx.x;
    if (i < n) out[i] = f2bf_bits(in[i]);
}

// Convert all 17 DxD weight matrices (15 time-block + 2 hypergraph) into
// transposed bf16 layout wt[w][n][k] so B fragments load contiguously in K.
__global__ void k_wt_prep(const float* __restrict__ tbW,
                          const float* __restrict__ hgW,
                          unsigned short* __restrict__ wt) {
    int i = blockIdx.x * blockDim.x + threadIdx.x;   // 17*128*128
    if (i >= 17 * DD * DD) return;
    int w = i >> 14; int r = i & 16383; int n = r >> 7; int k = r & 127;
    const float* src = (w < 15) ? (tbW + (size_t)w * DD * DD)
                                : (hgW + (size_t)(w - 15) * DD * DD);
    wt[i] = f2bf_bits(src[k * DD + n]);
}

__global__ void k_count_deg(const int* __restrict__ ni, const int* __restrict__ ei,
                            float* __restrict__ Dn, float* __restrict__ Be) {
    int i = blockIdx.x * blockDim.x + threadIdx.x;
    if (i >= NNZV) return;
    atomicAdd(&Dn[ni[i]], 1.0f);
    atomicAdd(&Be[ei[i]], 1.0f);
}

__global__ void k_recip(float* p, int n) {
    int i = blockIdx.x * blockDim.x + threadIdx.x;
    if (i < n) { float v = p[i]; p[i] = (v > 0.0f) ? (1.0f / v) : 0.0f; }
}

// ---- fused time block: relu(xW0+b0 + sigmoid(xW1+b1) + xW2+b2) --------------
// 64 rows per block, 8 waves x 16-col tiles, 4 row-tiles per wave.
// B fragments register-resident across row tiles (4x less weight traffic).

__global__ __launch_bounds__(256)
void k_timeblock(const unsigned short* __restrict__ act,   // [N][128] bf16
                 const unsigned short* __restrict__ wt,    // 3 mats, [n][k] bf16
                 const float* __restrict__ bias,           // [3][128] f32
                 unsigned short* __restrict__ out_bf,      // nullable
                 float* __restrict__ out_f32) {            // nullable
    __shared__ __align__(16) unsigned short tileA[64 * DD];   // 16 KB bf16
    const int row0 = blockIdx.x * 64;
    const int t = threadIdx.x;

    stage_tile_64x128(act + (size_t)row0 * DD, tileA, t);

    const int lane = t & 31, wave = t >> 5;
    const int g = lane >> 4, hn = lane & 15;
    const int ncol0 = wave * 16;

    v8f acc[4][3];
#pragma unroll
    for (int rt = 0; rt < 4; ++rt)
#pragma unroll
        for (int w = 0; w < 3; ++w) acc[rt][w] = (v8f){};

    const uint4* TA4 = (const uint4*)tileA;   // 16 uint4 per 128-elem row
    const uint4* WT4 = (const uint4*)wt;

#pragma unroll
    for (int kt = 0; kt < 4; ++kt) {
        // B fragments: K = 16*g + e, N = hn -> contiguous in transposed weights
        Frag16 b[3];
#pragma unroll
        for (int w = 0; w < 3; ++w) {
            size_t u4 = (((size_t)(w * DD + ncol0 + hn) * DD) + kt * 32 + g * 16) >> 3;
            uint4 p0 = WT4[u4], p1 = WT4[u4 + 1];
            b[w].u[0] = p0.x; b[w].u[1] = p0.y; b[w].u[2] = p0.z; b[w].u[3] = p0.w;
            b[w].u[4] = p1.x; b[w].u[5] = p1.y; b[w].u[6] = p1.z; b[w].u[7] = p1.w;
        }
#pragma unroll
        for (int rt = 0; rt < 4; ++rt) {
            // A fragment (K = 16*g + e, contiguous like B): two b128 LDS loads
            int rowi = rt * 16 + hn;
            Frag16 a;
            uint4 lo = TA4[rowi * 16 + kt * 4 + 2 * g];
            uint4 hi = TA4[rowi * 16 + kt * 4 + 2 * g + 1];
            a.u[0] = lo.x; a.u[1] = lo.y; a.u[2] = lo.z; a.u[3] = lo.w;
            a.u[4] = hi.x; a.u[5] = hi.y; a.u[6] = hi.z; a.u[7] = hi.w;
#pragma unroll
            for (int w = 0; w < 3; ++w)
                acc[rt][w] = __builtin_amdgcn_wmma_f32_16x16x32_bf16(
                    false, a.v, false, b[w].v, (short)0, acc[rt][w], false, false);
        }
    }

    // epilogue (fp32): relu(x0 + sigmoid(x1) + x2)
    const int col = ncol0 + hn;
    const float b0 = bias[col], b1 = bias[DD + col], b2 = bias[2 * DD + col];
#pragma unroll
    for (int rt = 0; rt < 4; ++rt) {
#pragma unroll
        for (int r = 0; r < 8; ++r) {
            int row = row0 + rt * 16 + g * 8 + r;      // C layout: M = r + 8*g
            float x0 = acc[rt][0][r] + b0;
            float x1 = acc[rt][1][r] + b1;
            float x2 = acc[rt][2][r] + b2;
            float s = 1.0f / (1.0f + __expf(-x1));
            float y = x0 + s + x2;
            y = y > 0.0f ? y : 0.0f;
            if (out_bf)  out_bf[(size_t)row * DD + col] = f2bf_bits(y);
            if (out_f32) out_f32[(size_t)row * DD + col] = y;
        }
    }
}

// ---------------- hypergraph-conv GEMM: xp = h @ W (fp32 out, no bias) -------

__global__ __launch_bounds__(256)
void k_gemm1(const unsigned short* __restrict__ act,
             const unsigned short* __restrict__ wt,   // [n][k] bf16
             float* __restrict__ out) {
    __shared__ __align__(16) unsigned short tileA[64 * DD];
    const int row0 = blockIdx.x * 64;
    const int t = threadIdx.x;

    stage_tile_64x128(act + (size_t)row0 * DD, tileA, t);

    const int lane = t & 31, wave = t >> 5;
    const int g = lane >> 4, hn = lane & 15;
    const int ncol0 = wave * 16;

    v8f acc[4];
#pragma unroll
    for (int rt = 0; rt < 4; ++rt) acc[rt] = (v8f){};

    const uint4* TA4 = (const uint4*)tileA;
    const uint4* WT4 = (const uint4*)wt;

#pragma unroll
    for (int kt = 0; kt < 4; ++kt) {
        Frag16 b;
        size_t u4 = (((size_t)(ncol0 + hn) * DD) + kt * 32 + g * 16) >> 3;
        uint4 p0 = WT4[u4], p1 = WT4[u4 + 1];
        b.u[0] = p0.x; b.u[1] = p0.y; b.u[2] = p0.z; b.u[3] = p0.w;
        b.u[4] = p1.x; b.u[5] = p1.y; b.u[6] = p1.z; b.u[7] = p1.w;
#pragma unroll
        for (int rt = 0; rt < 4; ++rt) {
            int rowi = rt * 16 + hn;
            Frag16 a;
            uint4 lo = TA4[rowi * 16 + kt * 4 + 2 * g];
            uint4 hi = TA4[rowi * 16 + kt * 4 + 2 * g + 1];
            a.u[0] = lo.x; a.u[1] = lo.y; a.u[2] = lo.z; a.u[3] = lo.w;
            a.u[4] = hi.x; a.u[5] = hi.y; a.u[6] = hi.z; a.u[7] = hi.w;
            acc[rt] = __builtin_amdgcn_wmma_f32_16x16x32_bf16(
                false, a.v, false, b.v, (short)0, acc[rt], false, false);
        }
    }

    const int col = ncol0 + hn;
#pragma unroll
    for (int rt = 0; rt < 4; ++rt)
#pragma unroll
        for (int r = 0; r < 8; ++r)
            out[(size_t)(row0 + rt * 16 + g * 8 + r) * DD + col] = acc[rt][r];
}

// ---------------- scatter phases (wave = one incidence entry, 128 cols) ------

__global__ __launch_bounds__(256)
void k_scatter_n2e(const float* __restrict__ xp, const int* __restrict__ ni,
                   const int* __restrict__ ei, float* __restrict__ m) {
    int gid = blockIdx.x * 256 + threadIdx.x;
    int entry = gid >> 5, c = gid & 31;
    if (entry >= NNZV) return;
    int node = ni[entry], edge = ei[entry];
    const float* src = xp + (size_t)node * DD + c;
    float* dst = m + (size_t)edge * DD + c;
#pragma unroll
    for (int j = 0; j < 4; ++j) atomicAdd(dst + 32 * j, src[32 * j]);
}

__global__ __launch_bounds__(256)
void k_scatter_e2n(const float* __restrict__ m, const float* __restrict__ Be,
                   const int* __restrict__ ni, const int* __restrict__ ei,
                   float* __restrict__ out) {
    int gid = blockIdx.x * 256 + threadIdx.x;
    int entry = gid >> 5, c = gid & 31;
    if (entry >= NNZV) return;
    int node = ni[entry], edge = ei[entry];
    float be = Be[edge];                           // fold 1/B_e into gather
    const float* src = m + (size_t)edge * DD + c;  // E*D = 512KB, L2-resident
    float* dst = out + (size_t)node * DD + c;
#pragma unroll
    for (int j = 0; j < 4; ++j) atomicAdd(dst + 32 * j, src[32 * j] * be);
}

__global__ void k_finalize(const float* __restrict__ acc, const float* __restrict__ Dn,
                           const float* __restrict__ bias,
                           unsigned short* __restrict__ out_bf) {
    int i = blockIdx.x * blockDim.x + threadIdx.x;   // N*128
    if (i >= NNODES * DD) return;
    int node = i >> 7, c = i & 127;
    out_bf[i] = f2bf_bits(acc[i] * Dn[node] + bias[c]);
}

// ---------------- host orchestration ----------------

extern "C" void kernel_launch(void* const* d_in, const int* in_sizes, int n_in,
                              void* d_out, int out_size, void* d_ws, size_t ws_size,
                              hipStream_t stream) {
    (void)in_sizes; (void)n_in; (void)out_size; (void)ws_size;
    const float* x    = (const float*)d_in[0];
    const float* tbW  = (const float*)d_in[1];   // [5][3][128][128]
    const float* tbB  = (const float*)d_in[2];   // [5][3][128]
    const float* hgW  = (const float*)d_in[3];   // [2][128][128]
    const float* hgB  = (const float*)d_in[4];   // [2][128]
    const int*   ni   = (const int*)d_in[6];
    const int*   ei   = (const int*)d_in[7];
    float* out = (float*)d_out;                  // also used as fp32 scratch (xp/acc)

    // workspace carve-out (~104.3 MB)
    char* ws = (char*)d_ws; size_t off = 0;
    auto carve = [&](size_t bytes) { char* p = ws + off; off += (bytes + 255) & ~(size_t)255; return p; };
    unsigned short* wt   = (unsigned short*)carve((size_t)17 * DD * DD * 2);
    float*          Dn   = (float*)carve((size_t)NNODES * 4);
    float*          Be   = (float*)carve((size_t)NEDGES * 4);
    float*          m    = (float*)carve((size_t)NEDGES * DD * 4);
    unsigned short* actA = (unsigned short*)carve((size_t)NNODES * DD * 2);
    unsigned short* actB = (unsigned short*)carve((size_t)NNODES * DD * 2);

    const int ND = NNODES * DD;                  // 25.6M
    const int gemmBlocks = NNODES / 64;          // 3125
    const int scatBlocks = (NNZV * 32) / 256;    // 200000

    // one-time prep
    k_wt_prep<<<(17 * DD * DD + 255) / 256, 256, 0, stream>>>(tbW, hgW, wt);
    k_zero_f32<<<(NNODES + 255) / 256, 256, 0, stream>>>(Dn, NNODES);
    k_zero_f32<<<(NEDGES + 255) / 256, 256, 0, stream>>>(Be, NEDGES);
    k_count_deg<<<(NNZV + 255) / 256, 256, 0, stream>>>(ni, ei, Dn, Be);
    k_recip<<<(NNODES + 255) / 256, 256, 0, stream>>>(Dn, NNODES);
    k_recip<<<(NEDGES + 255) / 256, 256, 0, stream>>>(Be, NEDGES);
    k_f32_to_bf16<<<(ND + 255) / 256, 256, 0, stream>>>(x, actA, ND);

    auto hconv = [&](const unsigned short* hin, int which, unsigned short* hout) {
        k_gemm1<<<gemmBlocks, 256, 0, stream>>>(hin, wt + (size_t)(15 + which) * DD * DD, out);
        k_zero_f32<<<(NEDGES * DD + 255) / 256, 256, 0, stream>>>(m, NEDGES * DD);
        k_scatter_n2e<<<scatBlocks, 256, 0, stream>>>(out, ni, ei, m);
        k_zero_f32<<<(ND + 255) / 256, 256, 0, stream>>>(out, ND);
        k_scatter_e2n<<<scatBlocks, 256, 0, stream>>>(m, Be, ni, ei, out);
        k_finalize<<<(ND + 255) / 256, 256, 0, stream>>>(out, Dn, hgB + which * DD, hout);
    };
    auto tblock = [&](const unsigned short* hin, int blk,
                      unsigned short* obf, float* of32) {
        k_timeblock<<<gemmBlocks, 256, 0, stream>>>(
            hin, wt + (size_t)blk * 3 * DD * DD, tbB + (size_t)blk * 3 * DD, obf, of32);
    };

    tblock(actA, 0, actB, nullptr);   // time block 0
    hconv(actB, 0, actA);             // hypergraph conv 0
    tblock(actA, 1, actB, nullptr);   // time block 1
    tblock(actB, 2, actA, nullptr);   // time block 2
    hconv(actA, 1, actB);             // hypergraph conv 1
    tblock(actB, 3, actA, nullptr);   // time block 3
    tblock(actA, 4, nullptr, out);    // time block 4 -> fp32 d_out
}
